// TransformerBlock_47708496724799
// MI455X (gfx1250) — matrix operand order repaired and checked by reading, not compiled, not run on hardware
//
#include <hip/hip_runtime.h>

// ---------------------------------------------------------------------------
// MI455X (gfx1250) transformer block, bf16-WMMA compute / f32 accumulate.
//   B=8, T=1024, C=1024, H=16, D=64, HIDDEN=4096
// GEMMs: double-buffered LDS tiles (128x128 x K-step 64), async-to-LDS copies
// (ASYNCcnt), bank-conflict-free padded LDS stride, v_wmma_f32_16x16x32_bf16.
// Attention: one wave / 16 query rows, global->WMMA streaming + LDS softmax.
// Workspace (bytes):
//   [0        ) wqkv_t  bf16 [3072][1024]   6 MiB
//   [6291456  ) wout_t  bf16 [1024][1024]   2 MiB
//   [8388608  ) wup_t   bf16 [4096][1024]   8 MiB
//   [16777216 ) wdown_t bf16 [1024][4096]   8 MiB
//   [25165824 ) q   bf16 [128][1024][64]   16 MiB  \
//   [41943040 ) k   bf16 [128][1024][64]   16 MiB   | reused as `up`
//   [58720256 ) vt  bf16 [128][64][1024]   16 MiB   | bf16 [8192][4096]
//   [75497472 ) h/y bf16 [8192][1024]      16 MiB  /
//   [92274688 ) x1  f32  [8192][1024]      32 MiB
//   [125829120) h2  bf16 [8192][1024]      16 MiB     total 136 MiB
// ---------------------------------------------------------------------------

#define USE_ASYNC_LDS 1

typedef __attribute__((ext_vector_type(16))) __bf16       v16bf;
typedef __attribute__((ext_vector_type(8)))  float        v8f;
typedef __attribute__((ext_vector_type(8)))  unsigned int v8u;
typedef unsigned short u16b;

__device__ __forceinline__ u16b f2bf(float f) {
  unsigned int u = __builtin_bit_cast(unsigned int, f);
  u += 0x7FFFu + ((u >> 16) & 1u);              // round-to-nearest-even
  return (u16b)(u >> 16);
}

__device__ __forceinline__ v16bf make_frag(uint4 lo, uint4 hi) {
  v8u u = {lo.x, lo.y, lo.z, lo.w, hi.x, hi.y, hi.z, hi.w};
  return __builtin_bit_cast(v16bf, u);
}

// A fragment, 16(M)x32(K) bf16, ISA 7.12.2: lane m=lane&15, hi=lane>>4 holds
// K runs [8*hi, 8*hi+8) and [16+8*hi, 16+8*hi+8)  -> two 16B loads per lane.
__device__ __forceinline__ v16bf load_a_frag(const u16b* rowptr, int hi) {
  const uint4* p0 = reinterpret_cast<const uint4*>(rowptr + 8 * hi);
  const uint4* p1 = reinterpret_cast<const uint4*>(rowptr + 16 + 8 * hi);
  return make_frag(*p0, *p1);
}

// B fragment, 32(K)x16(N) bf16 fed from an N-major (Bt[N][K]) row: lane
// n=lane&15, hi=lane>>4 holds K runs [8*hi, 8*hi+8) and [16+8*hi, 16+8*hi+8).
__device__ __forceinline__ v16bf load_b_frag(const u16b* rowptr, int hi) {
  const uint4* p0 = reinterpret_cast<const uint4*>(rowptr + 8 * hi);
  const uint4* p1 = reinterpret_cast<const uint4*>(rowptr + 16 + 8 * hi);
  return make_frag(*p0, *p1);
}

__device__ __forceinline__ v8f wmma_bf16(v16bf a, v16bf b, v8f c) {
  return __builtin_amdgcn_wmma_f32_16x16x32_bf16(false, a, false, b,
                                                 (short)0, c, false, false);
}

__device__ __forceinline__ v8f v8f_zero() {
  v8f z = {0.f, 0.f, 0.f, 0.f, 0.f, 0.f, 0.f, 0.f};
  return z;
}

// ---------------------------------------------------------------------------
// Async global->LDS copy (CDNA5, tracked by ASYNCcnt).  For flat pointers in
// the LDS aperture the low 32 bits are the wave-relative LDS byte offset
// (ISA 10.2), which is what VDST of global_load_async_to_lds expects.
// ---------------------------------------------------------------------------
__device__ __forceinline__ void async_copy_b128(void* lds_dst,
                                                const void* gsrc) {
#if USE_ASYNC_LDS
  unsigned lds_off = (unsigned)(unsigned long long)(uintptr_t)lds_dst;
  unsigned long long ga = (unsigned long long)(uintptr_t)gsrc;
  asm volatile("global_load_async_to_lds_b128 %0, %1, off"
               :: "v"(lds_off), "v"(ga)
               : "memory");
#else
  *reinterpret_cast<uint4*>(lds_dst) =
      *reinterpret_cast<const uint4*>(gsrc);
#endif
}

__device__ __forceinline__ void wait_async_copies() {
#if USE_ASYNC_LDS
  asm volatile("s_wait_asynccnt 0x0" ::: "memory");
#endif
}

// ---------------------------------------------------------------------------
// Weight conversion: f32 [K][N] row-major -> bf16 [N][K] row-major,
// LDS-tiled 32x32 so both global streams are coalesced.
// ---------------------------------------------------------------------------
__global__ __launch_bounds__(256) void transpose_f32_to_bf16(
    const float* __restrict__ in, u16b* __restrict__ out, int K, int N) {
  __shared__ float tile[32][33];
  const int tid = threadIdx.x;
  const int c = tid & 31, r = tid >> 5;                 // 32 x 8 threads
  const int n0 = blockIdx.x * 32, k0 = blockIdx.y * 32;
#pragma unroll
  for (int j = 0; j < 4; ++j)
    tile[r + j * 8][c] = in[(k0 + r + j * 8) * N + n0 + c];
  __syncthreads();
#pragma unroll
  for (int j = 0; j < 4; ++j)
    out[(n0 + r + j * 8) * K + k0 + c] = f2bf(tile[c][r + j * 8]);
}

// ---------------------------------------------------------------------------
// LayerNorm over C=1024, one 256-thread block per row, bf16 output.
// ---------------------------------------------------------------------------
__global__ __launch_bounds__(256) void layernorm_to_bf16(
    const float* __restrict__ x, const float* __restrict__ g,
    const float* __restrict__ beta, u16b* __restrict__ out) {
  __shared__ float s1[256];
  __shared__ float s2[256];
  const int row = blockIdx.x, tid = threadIdx.x;
  const float4 v4 = reinterpret_cast<const float4*>(x + row * 1024)[tid];
  float vv[4] = {v4.x, v4.y, v4.z, v4.w};
  s1[tid] = vv[0] + vv[1] + vv[2] + vv[3];
  s2[tid] = vv[0] * vv[0] + vv[1] * vv[1] + vv[2] * vv[2] + vv[3] * vv[3];
  __syncthreads();
  for (int off = 128; off > 0; off >>= 1) {
    if (tid < off) { s1[tid] += s1[tid + off]; s2[tid] += s2[tid + off]; }
    __syncthreads();
  }
  const float mean = s1[0] * (1.f / 1024.f);
  const float var  = s2[0] * (1.f / 1024.f) - mean * mean;
  const float inv  = rsqrtf(var + 1e-5f);
#pragma unroll
  for (int j = 0; j < 4; ++j) {
    const int c = tid * 4 + j;
    out[row * 1024 + c] = f2bf((vv[j] - mean) * inv * g[c] + beta[c]);
  }
}

// ---------------------------------------------------------------------------
// bf16 WMMA GEMM: C[M][N] = A[M][K] * Bt[N][K]^T + bias, templated epilogue.
//   256 threads = 8 waves (2x4), tile 128x128, K-step 64, double-buffered.
//   LDS row stride 88 u16 (176 B = 44-bank step): conflict-free b128 frags.
// MODE 0: QKV split -> q/k bf16 [B*H][T][D], vt bf16 [B*H][D][T]
// MODE 1: f32 out = acc + bias + res      (out projection / down projection)
// MODE 2: bf16 out = relu(acc + bias)     (up projection)
// ---------------------------------------------------------------------------
#define LDSS 88   // padded LDS row stride in u16 elements

template <int MODE>
__global__ __launch_bounds__(256) void gemm_bf16_128x128(
    const u16b* __restrict__ A, const u16b* __restrict__ Bt,
    const float* __restrict__ bias, const float* __restrict__ res,
    void* __restrict__ o0, void* __restrict__ o1, void* __restrict__ o2,
    int M, int N, int K) {
  __shared__ u16b sA[2][128 * LDSS];
  __shared__ u16b sB[2][128 * LDSS];
  const int tid = threadIdx.x;
  const int lane = tid & 31, wave = tid >> 5;
  const int lane16 = lane & 15, hi = lane >> 4;
  const int wm = wave & 1, wn = wave >> 1;              // 2 x 4 wave grid
  const int m0 = blockIdx.y * 128;
  const int n0 = blockIdx.x * 128;

  v8f acc[4][2];
#pragma unroll
  for (int mf = 0; mf < 4; ++mf)
#pragma unroll
    for (int nf = 0; nf < 2; ++nf) acc[mf][nf] = v8f_zero();

  // copy mapping: 128 rows x 64 k = 1024 x 16B chunks per matrix per step;
  // thread t owns chunks t*4 .. t*4+3 (contiguous 64 B of one row).
  const int crow = tid >> 1;                   // 2 threads per row
  const int cko  = (tid & 1) * 32;             // first of 4 8-elem chunks

  const int nsteps = K >> 6;                   // K-step 64

  auto copy_tiles = [&](int buf, int k0) {
    const u16b* ga = &A[(m0 + crow) * K + k0 + cko];
    const u16b* gb = &Bt[(n0 + crow) * K + k0 + cko];
    u16b* la = &sA[buf][crow * LDSS + cko];
    u16b* lb = &sB[buf][crow * LDSS + cko];
#pragma unroll
    for (int j = 0; j < 4; ++j) {
      async_copy_b128(la + j * 8, ga + j * 8);
      async_copy_b128(lb + j * 8, gb + j * 8);
    }
  };

  copy_tiles(0, 0);
  for (int s = 0; s < nsteps; ++s) {
    wait_async_copies();
    __syncthreads();                           // tile s resident in buf s&1
    if (s + 1 < nsteps) copy_tiles((s + 1) & 1, (s + 1) << 6);
    const u16b* curA = sA[s & 1];
    const u16b* curB = sB[s & 1];
#pragma unroll
    for (int w = 0; w < 2; ++w) {              // two 32-deep K windows
      v16bf bfr[2];
#pragma unroll
      for (int nf = 0; nf < 2; ++nf)
        bfr[nf] = load_b_frag(
            &curB[(wn * 32 + nf * 16 + lane16) * LDSS + w * 32], hi);
#pragma unroll
      for (int mf = 0; mf < 4; ++mf) {
        v16bf af = load_a_frag(
            &curA[(wm * 64 + mf * 16 + lane16) * LDSS + w * 32], hi);
        acc[mf][0] = wmma_bf16(af, bfr[0], acc[mf][0]);
        acc[mf][1] = wmma_bf16(af, bfr[1], acc[mf][1]);
      }
    }
  }

#pragma unroll
  for (int mf = 0; mf < 4; ++mf) {
#pragma unroll
    for (int nf = 0; nf < 2; ++nf) {
      const int c = n0 + wn * 32 + nf * 16 + lane16;
      const float bc = bias[c];
#pragma unroll
      for (int v = 0; v < 8; ++v) {
        const int r = m0 + wm * 64 + mf * 16 + v + 8 * hi;
        const float val = acc[mf][nf][v] + bc;
        if (MODE == 0) {
          const int grp = c >> 10, cc = c & 1023;       // uniform across wave
          const int h = cc >> 6, d = cc & 63;
          const int bh = ((r >> 10) << 4) + h, t = r & 1023;
          if (grp == 0)
            ((u16b*)o0)[(bh * 1024 + t) * 64 + d] = f2bf(val);
          else if (grp == 1)
            ((u16b*)o1)[(bh * 1024 + t) * 64 + d] = f2bf(val);
          else
            ((u16b*)o2)[(bh * 64 + d) * 1024 + t] = f2bf(val);
        } else if (MODE == 1) {
          ((float*)o0)[r * N + c] = val + res[r * N + c];
        } else {
          ((u16b*)o0)[r * N + c] = f2bf(fmaxf(val, 0.f));
        }
      }
    }
  }
}

// ---------------------------------------------------------------------------
// Attention: softmax(Q K^T / sqrt(D)) V per (b,h).  One wave (32 threads) per
// block handles 16 query rows; 64 KB LDS score strip; zero barriers needed
// (wave-private rows), WMMA runs with full EXEC as the ISA requires.
//   grid = (T/16 = 64, B*H = 128)
// ---------------------------------------------------------------------------
__global__ __launch_bounds__(32) void attention_kernel(
    const u16b* __restrict__ q, const u16b* __restrict__ k,
    const u16b* __restrict__ vt, u16b* __restrict__ y) {
  __shared__ float S[16 * 1024];                        // 64 KiB
  const int qt = blockIdx.x;                            // query tile
  const int bh = blockIdx.y;                            // batch*head
  const int lane = threadIdx.x;
  const int lane16 = lane & 15, hi = lane >> 4;
  const float scale = 0.125f;                           // 1/sqrt(64)

  // Q fragments for this 16x64 tile (K windows 0 and 1 of D=64)
  const u16b* qrow = q + ((bh << 10) + (qt << 4) + lane16) * 64;
  const v16bf aq0 = load_a_frag(qrow, hi);
  const v16bf aq1 = load_a_frag(qrow + 32, hi);

  // ---- scores: S[16][1024] = scale * Q K^T  (K[T][D] is already "Bt") ----
  for (int nt = 0; nt < 64; ++nt) {
    const u16b* krow = k + ((bh << 10) + (nt << 4) + lane16) * 64;
    v8f accs = v8f_zero();
    accs = wmma_bf16(aq0, load_b_frag(krow, hi), accs);
    accs = wmma_bf16(aq1, load_b_frag(krow + 32, hi), accs);
#pragma unroll
    for (int v = 0; v < 8; ++v)
      S[(v + 8 * hi) * 1024 + (nt << 4) + lane16] = accs[v] * scale;
  }
  __syncthreads();  // single-wave WG: trivial, keeps LDS ordering explicit

  // ---- softmax rows, write bf16 probabilities in place ----
  for (int rr = 0; rr < 16; ++rr) {
    float* Sr = S + rr * 1024;
    float mx = -3.402823466e38f;
    for (int i = 0; i < 32; ++i) mx = fmaxf(mx, Sr[i * 32 + lane]);
    for (int off = 16; off > 0; off >>= 1)
      mx = fmaxf(mx, __shfl_xor(mx, off, 32));
    float sm = 0.f;
    for (int i = 0; i < 32; ++i) sm += __expf(Sr[i * 32 + lane] - mx);
    for (int off = 16; off > 0; off >>= 1) sm += __shfl_xor(sm, off, 32);
    const float inv = 1.f / sm;
    u16b* Pr = (u16b*)Sr;              // bf16 row overlays first half of f32 row
    const float pre = Sr[lane];        // pre-read iter-0 values before any store
    for (int i = 0; i < 32; ++i) {
      const int c = i * 32 + lane;
      const float sv = i ? Sr[c] : pre;
      Pr[c] = f2bf(__expf(sv - mx) * inv);
    }
  }
  __syncthreads();

  // ---- O[16][64] = P[16][1024] @ V ;  vt[D][T] is "Bt" for this GEMM ----
  v8f acco[4];
#pragma unroll
  for (int nf = 0; nf < 4; ++nf) acco[nf] = v8f_zero();
  const u16b* Prow = (const u16b*)(S + lane16 * 1024);
  for (int kk = 0; kk < 32; ++kk) {
    const v16bf af = load_a_frag(Prow + kk * 32, hi);
#pragma unroll
    for (int nf = 0; nf < 4; ++nf) {
      const u16b* vrow = vt + ((bh * 64 + nf * 16 + lane16) << 10);
      acco[nf] = wmma_bf16(af, load_b_frag(vrow + kk * 32, hi), acco[nf]);
    }
  }
  const int bb = bh >> 4, hh = bh & 15;
#pragma unroll
  for (int nf = 0; nf < 4; ++nf)
#pragma unroll
    for (int v = 0; v < 8; ++v) {
      const int tq = (qt << 4) + v + 8 * hi;
      const int col = hh * 64 + nf * 16 + lane16;
      y[((bb << 10) + tq) * 1024 + col] = f2bf(acco[nf][v]);
    }
}

// ---------------------------------------------------------------------------
extern "C" void kernel_launch(void* const* d_in, const int* in_sizes, int n_in,
                              void* d_out, int out_size, void* d_ws,
                              size_t ws_size, hipStream_t stream) {
  (void)in_sizes; (void)n_in; (void)out_size; (void)ws_size;
  const float* x      = (const float*)d_in[0];
  const float* ln1_g  = (const float*)d_in[1];
  const float* ln1_b  = (const float*)d_in[2];
  const float* w_qkv  = (const float*)d_in[3];
  const float* b_qkv  = (const float*)d_in[4];
  const float* w_out  = (const float*)d_in[5];
  const float* b_out  = (const float*)d_in[6];
  const float* ln2_g  = (const float*)d_in[7];
  const float* ln2_b  = (const float*)d_in[8];
  const float* w_up   = (const float*)d_in[9];
  const float* b_up   = (const float*)d_in[10];
  const float* w_down = (const float*)d_in[11];
  const float* b_down = (const float*)d_in[12];
  float* out = (float*)d_out;

  char* ws = (char*)d_ws;
  u16b*  wqkv_t  = (u16b*)(ws + 0);
  u16b*  wout_t  = (u16b*)(ws + 6291456);
  u16b*  wup_t   = (u16b*)(ws + 8388608);
  u16b*  wdown_t = (u16b*)(ws + 16777216);
  u16b*  qb      = (u16b*)(ws + 25165824);
  u16b*  kb      = (u16b*)(ws + 41943040);
  u16b*  vtb     = (u16b*)(ws + 58720256);
  u16b*  hb      = (u16b*)(ws + 75497472);   // LN1 out, then attention out y
  float* x1      = (float*)(ws + 92274688);
  u16b*  h2      = (u16b*)(ws + 125829120);
  u16b*  up      = (u16b*)(ws + 25165824);   // reuse q/k/vt/h region (64 MiB)

  // 1) weights -> bf16, transposed to [N][K] (tiled, coalesced both ways)
  transpose_f32_to_bf16<<<dim3(3072 / 32, 1024 / 32), 256, 0, stream>>>(w_qkv,  wqkv_t,  1024, 3072);
  transpose_f32_to_bf16<<<dim3(1024 / 32, 1024 / 32), 256, 0, stream>>>(w_out,  wout_t,  1024, 1024);
  transpose_f32_to_bf16<<<dim3(4096 / 32, 1024 / 32), 256, 0, stream>>>(w_up,   wup_t,   1024, 4096);
  transpose_f32_to_bf16<<<dim3(1024 / 32, 4096 / 32), 256, 0, stream>>>(w_down, wdown_t, 4096, 1024);

  // 2) h = LN1(x) in bf16
  layernorm_to_bf16<<<8192, 256, 0, stream>>>(x, ln1_g, ln1_b, hb);

  // 3) qkv = h @ w_qkv + b_qkv, split into q/k (row-major) and v (transposed)
  gemm_bf16_128x128<0><<<dim3(24, 64), 256, 0, stream>>>(
      hb, wqkv_t, b_qkv, nullptr, qb, kb, vtb, 8192, 3072, 1024);

  // 4) y = softmax(q k^T / sqrt(D)) v   (writes into hb)
  attention_kernel<<<dim3(64, 128), 32, 0, stream>>>(qb, kb, vtb, hb);

  // 5) x1 = x + y @ w_out + b_out
  gemm_bf16_128x128<1><<<dim3(8, 64), 256, 0, stream>>>(
      hb, wout_t, b_out, x, x1, nullptr, nullptr, 8192, 1024, 1024);

  // 6) h2 = LN2(x1)
  layernorm_to_bf16<<<8192, 256, 0, stream>>>(x1, ln2_g, ln2_b, h2);

  // 7) up = relu(h2 @ w_up + b_up)
  gemm_bf16_128x128<2><<<dim3(32, 64), 256, 0, stream>>>(
      h2, wup_t, b_up, nullptr, up, nullptr, nullptr, 8192, 4096, 1024);

  // 8) out = x1 + up @ w_down + b_down
  gemm_bf16_128x128<1><<<dim3(8, 64), 256, 0, stream>>>(
      up, wdown_t, b_down, x1, out, nullptr, nullptr, 8192, 1024, 4096);
}
